// LSTMCell_33036888441242
// MI455X (gfx1250) — matrix operand order repaired and checked by reading, not compiled, not run on hardware
//
#include <hip/hip_runtime.h>

// LSTM cell, B=4096, IN=1024, H=2048, fused single-pass formulation:
//   gates = [prev_h | x] @ [W_i;W_f;W_o;W_c]^T   (K = 3072, N = 4*2048)
// fp32 accuracy recovered via split-bf16 (hi+lo) WMMA emulation:
//   a*b ~= ah*bh + ah*bl + al*bh  (3x v_wmma_f32_16x16x32_bf16, f32 accum)
//
// Workspace layout (needs ~151 MB; fits entirely in the 192MB L2 during GEMM):
//   hx_hi [4096x3072] bf16 | hx_lo | w_hi [8192x3072] bf16 | w_lo

#define BDIM  4096
#define HDIM  2048
#define INDIM 1024
#define KTOT  3072   // HDIM + INDIM

typedef __attribute__((ext_vector_type(16))) __bf16 v16bf;
typedef __attribute__((ext_vector_type(8)))  __bf16 v8bf;
typedef __attribute__((ext_vector_type(8)))  float  v8f;

__device__ __forceinline__ __bf16 f32_to_bf16_rn(float f) {
  union { float f; unsigned u; } x; x.f = f;
  unsigned r = x.u + 0x7FFFu + ((x.u >> 16) & 1u);   // round-to-nearest-even
  union { unsigned short s; __bf16 b; } y; y.s = (unsigned short)(r >> 16);
  return y.b;
}
__device__ __forceinline__ float bf16_to_f32(__bf16 b) {
  union { unsigned short s; __bf16 b; } y; y.b = b;
  union { float f; unsigned u; } x; x.u = ((unsigned)y.s) << 16;
  return x.f;
}
__device__ __forceinline__ float fast_sigmoid(float x) {
  return 1.0f / (1.0f + __expf(-x));
}
__device__ __forceinline__ float fast_tanh(float x) {
  // safe at +/-inf: e->inf gives 1, e->0 gives -1
  float e = __expf(2.0f * x);
  return 1.0f - 2.0f / (e + 1.0f);
}

// Load a 16-element bf16 fragment as two contiguous 16B chunks (p, p+step).
__device__ __forceinline__ v16bf load_frag(const __bf16* __restrict__ p, int step) {
  union { v16bf w; v8bf h[2]; } u;
  u.h[0] = *(const v8bf*)(p);
  u.h[1] = *(const v8bf*)(p + step);
  return u.w;
}

// ---------------- split pre-pass: hx = [prev_h | x] -> bf16 hi/lo ----------------
__global__ __launch_bounds__(256) void lstm_split_hx(
    const float* __restrict__ x, const float* __restrict__ prev_h,
    __bf16* __restrict__ hx_hi, __bf16* __restrict__ hx_lo) {
  const int row = blockIdx.y;                       // 0..4095
  const int k   = blockIdx.x * 256 + threadIdx.x;   // 0..3071
  float v = (k < HDIM) ? prev_h[(size_t)row * HDIM + k]
                       : x[(size_t)row * INDIM + (k - HDIM)];
  __bf16 hi = f32_to_bf16_rn(v);
  __bf16 lo = f32_to_bf16_rn(v - bf16_to_f32(hi));
  size_t idx = (size_t)row * KTOT + k;
  hx_hi[idx] = hi;
  hx_lo[idx] = lo;
}

// ---------------- split pre-pass: W = [Wi;Wf;Wo;Wc] -> bf16 hi/lo ----------------
__global__ __launch_bounds__(256) void lstm_split_w(
    const float* __restrict__ Wi, const float* __restrict__ Wf,
    const float* __restrict__ Wo, const float* __restrict__ Wc,
    __bf16* __restrict__ w_hi, __bf16* __restrict__ w_lo) {
  const int rw = blockIdx.y;                        // 0..8191 (gate-stacked row)
  const int k  = blockIdx.x * 256 + threadIdx.x;    // 0..3071
  const int g  = rw >> 11;                          // rw / 2048
  const int r  = rw & 2047;
  const float* W = (g == 0) ? Wi : (g == 1) ? Wf : (g == 2) ? Wo : Wc;
  float v = W[(size_t)r * KTOT + k];
  __bf16 hi = f32_to_bf16_rn(v);
  __bf16 lo = f32_to_bf16_rn(v - bf16_to_f32(hi));
  size_t idx = (size_t)rw * KTOT + k;
  w_hi[idx] = hi;
  w_lo[idx] = lo;
}

// ---------------- fused WMMA GEMM + LSTM epilogue ----------------
// Block: 256 threads = 8 waves, 2(M) x 4(N). Block tile: 128 batch rows x 64 hidden cols.
// Each wave: 64x16 output tile for ALL 4 gates (16 accumulators), K-loop step 32.
__global__ __launch_bounds__(256) void lstm_wmma_gemm(
    const __bf16* __restrict__ hx_hi, const __bf16* __restrict__ hx_lo,
    const __bf16* __restrict__ w_hi,  const __bf16* __restrict__ w_lo,
    const float*  __restrict__ prev_c, float* __restrict__ out) {
  const int lane  = threadIdx.x & 31;
  const int wave  = threadIdx.x >> 5;
  const int waveM = wave >> 2;          // 0..1
  const int waveN = wave & 3;           // 0..3
  const int lh    = lane & 15;
  const int sel   = lane >> 4;          // 0 or 1 (half-wave select)

  const int rowBase = blockIdx.y * 128 + waveM * 64;  // batch row base
  const int colBase = blockIdx.x * 64  + waveN * 16;  // hidden col base

  // accumulators: [m-subtile][gate] = 16 x v8f
  v8f acc[4][4];
  #pragma unroll
  for (int mi = 0; mi < 4; ++mi)
    #pragma unroll
    for (int g = 0; g < 4; ++g)
      #pragma unroll
      for (int r = 0; r < 8; ++r) acc[mi][g][r] = 0.0f;

  // A fragment (16x32 bf16): lane = M row; lanes 0-15 hold K 0-7 & 16-23,
  // lanes 16-31 hold K 8-15 & 24-31  -> base +sel*8, chunks at +0 and +16.
  const size_t aOff = (size_t)(rowBase + lh) * KTOT + sel * 8;
  const __bf16* aH = hx_hi + aOff;
  const __bf16* aL = hx_lo + aOff;

  // B fragment (32x16 bf16): lane = N col (W row); lanes 0-15 hold K 0-7 & 16-23,
  // lanes 16-31 hold K 8-15 & 24-31 -> base +sel*8, chunks at +0 and +16.
  size_t bOff[4];
  #pragma unroll
  for (int g = 0; g < 4; ++g)
    bOff[g] = (size_t)(g * HDIM + colBase + lh) * KTOT + sel * 8;

  for (int k0 = 0; k0 < KTOT; k0 += 32) {
    v16bf Ah[4], Al[4];
    #pragma unroll
    for (int mi = 0; mi < 4; ++mi) {
      const size_t o = (size_t)mi * 16 * KTOT + k0;
      Ah[mi] = load_frag(aH + o, 16);
      Al[mi] = load_frag(aL + o, 16);
    }
    #pragma unroll
    for (int g = 0; g < 4; ++g) {
      v16bf Bh = load_frag(w_hi + bOff[g] + k0, 16);
      v16bf Bl = load_frag(w_lo + bOff[g] + k0, 16);
      #pragma unroll
      for (int mi = 0; mi < 4; ++mi) {
        acc[mi][g] = __builtin_amdgcn_wmma_f32_16x16x32_bf16(
            false, Ah[mi], false, Bh, (short)0, acc[mi][g], false, false);
        acc[mi][g] = __builtin_amdgcn_wmma_f32_16x16x32_bf16(
            false, Ah[mi], false, Bl, (short)0, acc[mi][g], false, false);
        acc[mi][g] = __builtin_amdgcn_wmma_f32_16x16x32_bf16(
            false, Al[mi], false, Bh, (short)0, acc[mi][g], false, false);
      }
    }
  }

  // Epilogue. C/D layout: VGPR r, lanes 0-15 -> M=r, lanes 16-31 -> M=8+r; N=lane%16.
  const int hcol = colBase + lh;
  const size_t outC = (size_t)BDIM * HDIM;  // next_c offset in d_out
  #pragma unroll
  for (int mi = 0; mi < 4; ++mi) {
    #pragma unroll
    for (int r = 0; r < 8; ++r) {
      const int brow = rowBase + mi * 16 + sel * 8 + r;
      const size_t o = (size_t)brow * HDIM + hcol;
      float pc = prev_c[o];
      float gi = acc[mi][0][r];
      float gf = acc[mi][1][r];
      float go = acc[mi][2][r];
      float gc = acc[mi][3][r];
      float i  = fast_sigmoid(gi);
      float f  = fast_sigmoid(gf);
      float ofl= fast_sigmoid(go);
      float ct = fast_tanh(gc);
      float nc = f * pc + i * ct;
      float nh = ofl * fast_tanh(nc);
      out[o]        = nh;   // next_h
      out[outC + o] = nc;   // next_c
    }
  }
}

extern "C" void kernel_launch(void* const* d_in, const int* in_sizes, int n_in,
                              void* d_out, int out_size, void* d_ws, size_t ws_size,
                              hipStream_t stream) {
  (void)in_sizes; (void)n_in; (void)out_size; (void)ws_size;
  const float* x      = (const float*)d_in[0];
  const float* prev_h = (const float*)d_in[1];
  const float* prev_c = (const float*)d_in[2];
  const float* Wi     = (const float*)d_in[3];
  const float* Wf     = (const float*)d_in[4];
  const float* Wo     = (const float*)d_in[5];
  const float* Wc     = (const float*)d_in[6];
  float* out = (float*)d_out;

  char* ws = (char*)d_ws;
  const size_t hxBytes = (size_t)BDIM * KTOT * sizeof(__bf16);      // 25.2 MB
  const size_t wBytes  = (size_t)4 * HDIM * KTOT * sizeof(__bf16);  // 50.3 MB
  __bf16* hx_hi = (__bf16*)(ws);
  __bf16* hx_lo = (__bf16*)(ws + hxBytes);
  __bf16* w_hi  = (__bf16*)(ws + 2 * hxBytes);
  __bf16* w_lo  = (__bf16*)(ws + 2 * hxBytes + wBytes);

  // split pre-passes (memory-bound, ~13us total)
  lstm_split_hx<<<dim3(KTOT / 256, BDIM), 256, 0, stream>>>(x, prev_h, hx_hi, hx_lo);
  lstm_split_w<<<dim3(KTOT / 256, 4 * HDIM), 256, 0, stream>>>(Wi, Wf, Wo, Wc, w_hi, w_lo);

  // fused GEMM + LSTM gates (compute-bound on WMMA)
  dim3 grid(HDIM / 64, BDIM / 128);  // 32 x 32 workgroups
  lstm_wmma_gemm<<<grid, 256, 0, stream>>>(hx_hi, hx_lo, w_hi, w_lo, prev_c, out);
}
